// LMModel_22497038697100
// MI455X (gfx1250) — compile-verified, matrix operand order checked by reading, not run on hardware
//
#include <hip/hip_runtime.h>
#include <hip/hip_bf16.h>
#include <math.h>
#include <stdint.h>

// ---------------- model dims ----------------
#define BDIM 4
#define SDIM 512
#define EDIM 768
#define HDIM 12
#define LDIM 12
#define VDIM 40990
#define FFDIM (4 * 768)
#define DH (EDIM / HDIM)          // 64
#define MROWS (BDIM * SDIM)       // 2048

typedef __attribute__((ext_vector_type(16))) __bf16 v16bf;
typedef __attribute__((ext_vector_type(8)))  float  v8f;

// f32 -> bf16 round-to-nearest-even
__device__ __forceinline__ unsigned short f2bf(float f) {
    unsigned int u = __float_as_uint(f);
    unsigned int lsb = (u >> 16) & 1u;
    u += 0x7FFFu + lsb;
    return (unsigned short)(u >> 16);
}

// Load a 16x32 bf16 WMMA fragment from LDS.
// ISA layout (16-bit A 16x32): lane l -> row m = l&15, half = l>>4;
// VGPR0-3 hold K = half*8 .. half*8+7 (contiguous 16B), VGPR4-7 hold K = 16+half*8 .. (+7).
__device__ __forceinline__ v16bf ldfrag(const unsigned short* p0, const unsigned short* p1) {
    union U { uint4 q[2]; v16bf v; } u;
    u.q[0] = *reinterpret_cast<const uint4*>(p0);
    u.q[1] = *reinterpret_cast<const uint4*>(p1);
    return u.v;
}

__device__ __forceinline__ v8f wmma_bf16(v16bf a, v16bf b, v8f c) {
    return __builtin_amdgcn_wmma_f32_16x16x32_bf16(false, a, false, b, (short)0, c, false, false);
}

__device__ __forceinline__ float gelu_f(float x) {
    float x3 = x * x * x;
    return 0.5f * x * (1.0f + tanhf(0.7978845608028654f * (x + 0.044715f * x3)));
}

// ---------------- embedding (async global->LDS gather) ----------------
// Each block handles one token row: three 768-float rows are DMA'd into LDS with
// GLOBAL_LOAD_ASYNC_TO_LDS_B128 (ASYNCcnt path), each wave waits its own asynccnt,
// barrier publishes, then the sum is written out. 192 threads * 4 floats = 768.
__global__ __launch_bounds__(192) void embed_kernel(const int* __restrict__ x,
                                                    const int* __restrict__ mask,
                                                    const float* __restrict__ embed_W,
                                                    const float* __restrict__ art_W,
                                                    float* __restrict__ h) {
    __shared__ __attribute__((aligned(16))) float E0[EDIM];
    __shared__ __attribute__((aligned(16))) float E1[EDIM];
    __shared__ __attribute__((aligned(16))) float D0[EDIM];

    const int row = blockIdx.x;           // b*S + s  (S=512 <= 514 -> always art_W)
    const int t   = threadIdx.x;
    const int x0 = x[row * 2 + 0];
    const int x1 = x[row * 2 + 1];
    const int mv = mask[row];

    const int e = t * 4;
    unsigned long long g0 = (unsigned long long)(uintptr_t)(embed_W + (size_t)x0 * EDIM + e);
    unsigned long long g1 = (unsigned long long)(uintptr_t)(embed_W + (size_t)x1 * EDIM + e);
    unsigned long long g2 = (unsigned long long)(uintptr_t)(art_W   + (size_t)mv * EDIM + e);
    unsigned l0 = (unsigned)(uintptr_t)&E0[e];   // low 32 bits of generic LDS ptr = LDS addr
    unsigned l1 = (unsigned)(uintptr_t)&E1[e];
    unsigned l2 = (unsigned)(uintptr_t)&D0[e];

    asm volatile("global_load_async_to_lds_b128 %0, %1, off" :: "v"(l0), "v"(g0) : "memory");
    asm volatile("global_load_async_to_lds_b128 %0, %1, off" :: "v"(l1), "v"(g1) : "memory");
    asm volatile("global_load_async_to_lds_b128 %0, %1, off" :: "v"(l2), "v"(g2) : "memory");
    asm volatile("s_wait_asynccnt 0" ::: "memory");
    __syncthreads();

    float4 a = *reinterpret_cast<const float4*>(&E0[e]);
    float4 b = *reinterpret_cast<const float4*>(&E1[e]);
    float4 c = *reinterpret_cast<const float4*>(&D0[e]);
    float4 o;
    o.x = a.x + b.x + c.x; o.y = a.y + b.y + c.y;
    o.z = a.z + b.z + c.z; o.w = a.w + b.w + c.w;
    *reinterpret_cast<float4*>(&h[(size_t)row * EDIM + e]) = o;
}

// ---------------- layernorm ----------------
__global__ void ln_kernel(const float* __restrict__ x, const float* __restrict__ g,
                          const float* __restrict__ b, float* __restrict__ y) {
    __shared__ float s1[256];
    __shared__ float s2[256];
    int row = blockIdx.x, t = threadIdx.x;
    const float* xr = x + (size_t)row * EDIM;
    float a0 = 0.f, a1 = 0.f;
    for (int e = t; e < EDIM; e += 256) { float v = xr[e]; a0 += v; a1 += v * v; }
    s1[t] = a0; s2[t] = a1;
    __syncthreads();
    for (int off = 128; off > 0; off >>= 1) {
        if (t < off) { s1[t] += s1[t + off]; s2[t] += s2[t + off]; }
        __syncthreads();
    }
    float mean = s1[0] * (1.0f / EDIM);
    float var  = s2[0] * (1.0f / EDIM) - mean * mean;
    float rstd = rsqrtf(var + 1e-5f);
    for (int e = t; e < EDIM; e += 256)
        y[(size_t)row * EDIM + e] = g[e] * (xr[e] - mean) * rstd + b[e];
}

// ---------------- GEMM: C[M,N] = A[M,K] @ B + epilogue ----------------
// BT=false: B is [K,N] row-major.  BT=true: B is [N,K] row-major (used for h @ embed_W^T).
// EPI: 0 none, 1 +bias, 2 +bias then GELU, 3 +bias +residual
// 128x128 tile, BK=32, 8 waves (wave grid 4m x 2n; each wave 32m x 64n = 8 WMMA/step),
// double-buffered LDS staging + global prefetch of the k+2 tile.
template<int EPI, bool BT>
__global__ __launch_bounds__(256, 1) void gemm_kernel(
        const float* __restrict__ A, const float* __restrict__ Bm,
        const float* __restrict__ bias, const float* __restrict__ res,
        float* __restrict__ C, int M, int N, int K) {
    constexpr int BM = 128, BN = 128, BK = 32, LDT = 40;  // 40*2B = 80B row stride (16B aligned)
    __shared__ __attribute__((aligned(16))) unsigned short As[2][BM * LDT];
    __shared__ __attribute__((aligned(16))) unsigned short Bs[2][BN * LDT];

    const int tid  = threadIdx.x;
    const int n0   = blockIdx.x * BN;
    const int m0   = blockIdx.y * BM;
    const int lane = tid & 31;
    const int w    = tid >> 5;           // 8 waves
    const int wm   = (w & 3) * 32;       // wave row offset in tile (4 waves along M)
    const int wn   = (w >> 2) * 64;      // wave col offset in tile (2 waves along N)
    const int half = lane >> 4;
    const int l15  = lane & 15;

    const v8f vz = {0.f, 0.f, 0.f, 0.f, 0.f, 0.f, 0.f, 0.f};
    v8f acc[2][4];
    #pragma unroll
    for (int i = 0; i < 2; ++i)
        #pragma unroll
        for (int j = 0; j < 4; ++j) acc[i][j] = vz;

    auto stage = [&](int k0, int pb) {
        // ---- A tile 128x32 (f32 -> bf16); thread: row=tid/2, 16 cols ----
        {
            int row = tid >> 1, cg = (tid & 1) * 16;
            const float* gp = A + (size_t)(m0 + row) * K + k0 + cg;
            float4 f0 = *reinterpret_cast<const float4*>(gp);
            float4 f1 = *reinterpret_cast<const float4*>(gp + 4);
            float4 f2 = *reinterpret_cast<const float4*>(gp + 8);
            float4 f3 = *reinterpret_cast<const float4*>(gp + 12);
            unsigned short* d = &As[pb][row * LDT + cg];
            d[0]  = f2bf(f0.x); d[1]  = f2bf(f0.y); d[2]  = f2bf(f0.z); d[3]  = f2bf(f0.w);
            d[4]  = f2bf(f1.x); d[5]  = f2bf(f1.y); d[6]  = f2bf(f1.z); d[7]  = f2bf(f1.w);
            d[8]  = f2bf(f2.x); d[9]  = f2bf(f2.y); d[10] = f2bf(f2.z); d[11] = f2bf(f2.w);
            d[12] = f2bf(f3.x); d[13] = f2bf(f3.y); d[14] = f2bf(f3.z); d[15] = f2bf(f3.w);
        }
        // ---- B tile as Bs[n][k] ----
        if (BT) {
            int row = tid >> 1, cg = (tid & 1) * 16;   // row = n, cg = k group
            float4 f0 = {0.f,0.f,0.f,0.f}, f1 = f0, f2 = f0, f3 = f0;
            if (n0 + row < N) {
                const float* gp = Bm + (size_t)(n0 + row) * K + k0 + cg;
                f0 = *reinterpret_cast<const float4*>(gp);
                f1 = *reinterpret_cast<const float4*>(gp + 4);
                f2 = *reinterpret_cast<const float4*>(gp + 8);
                f3 = *reinterpret_cast<const float4*>(gp + 12);
            }
            unsigned short* d = &Bs[pb][row * LDT + cg];
            d[0]  = f2bf(f0.x); d[1]  = f2bf(f0.y); d[2]  = f2bf(f0.z); d[3]  = f2bf(f0.w);
            d[4]  = f2bf(f1.x); d[5]  = f2bf(f1.y); d[6]  = f2bf(f1.z); d[7]  = f2bf(f1.w);
            d[8]  = f2bf(f2.x); d[9]  = f2bf(f2.y); d[10] = f2bf(f2.z); d[11] = f2bf(f2.w);
            d[12] = f2bf(f3.x); d[13] = f2bf(f3.y); d[14] = f2bf(f3.z); d[15] = f2bf(f3.w);
        } else {
            int kk = tid >> 3, ng = (tid & 7) * 16;    // kk = k row, ng = n group (N % 128 == 0 here)
            const float* gp = Bm + (size_t)(k0 + kk) * N + n0 + ng;
            float4 f0 = *reinterpret_cast<const float4*>(gp);
            float4 f1 = *reinterpret_cast<const float4*>(gp + 4);
            float4 f2 = *reinterpret_cast<const float4*>(gp + 8);
            float4 f3 = *reinterpret_cast<const float4*>(gp + 12);
            float v[16] = {f0.x, f0.y, f0.z, f0.w, f1.x, f1.y, f1.z, f1.w,
                           f2.x, f2.y, f2.z, f2.w, f3.x, f3.y, f3.z, f3.w};
            #pragma unroll
            for (int i = 0; i < 16; ++i) Bs[pb][(ng + i) * LDT + kk] = f2bf(v[i]);
        }
    };

    const int NS = K / BK;
    stage(0, 0);
    __syncthreads();

    for (int ks = 0; ks < NS; ++ks) {
        const int pb = ks & 1;
        if (ks + 1 < NS) stage((ks + 1) * BK, pb ^ 1);
        if (ks + 2 < NS) {   // software prefetch of k+2 tile -> global_prefetch path
            __builtin_prefetch(A + (size_t)(m0 + (tid >> 1)) * K + (ks + 2) * BK + (tid & 1) * 16, 0, 1);
            if (!BT) __builtin_prefetch(Bm + (size_t)((ks + 2) * BK + (tid >> 3)) * N + n0 + (tid & 7) * 16, 0, 1);
        }

        v16bf af0 = ldfrag(&As[pb][(wm      + l15) * LDT + half * 8],
                           &As[pb][(wm      + l15) * LDT + 16 + half * 8]);
        v16bf af1 = ldfrag(&As[pb][(wm + 16 + l15) * LDT + half * 8],
                           &As[pb][(wm + 16 + l15) * LDT + 16 + half * 8]);
        #pragma unroll
        for (int nf = 0; nf < 4; ++nf) {
            v16bf bfv = ldfrag(&Bs[pb][(wn + nf * 16 + l15) * LDT + half * 8],
                               &Bs[pb][(wn + nf * 16 + l15) * LDT + 16 + half * 8]);
            acc[0][nf] = wmma_bf16(af0, bfv, acc[0][nf]);
            acc[1][nf] = wmma_bf16(af1, bfv, acc[1][nf]);
        }
        __syncthreads();
    }

    // ---- store with epilogue (C/D layout: VGPR r -> row (half*8+r), col = l15) ----
    #pragma unroll
    for (int mf = 0; mf < 2; ++mf) {
        #pragma unroll
        for (int r = 0; r < 8; ++r) {
            int row = m0 + wm + mf * 16 + half * 8 + r;
            #pragma unroll
            for (int nf = 0; nf < 4; ++nf) {
                int col = n0 + wn + nf * 16 + l15;
                if (BT && col >= N) continue;
                float v = acc[mf][nf][r];
                if (EPI >= 1) v += bias[col];
                if (EPI == 2) v = gelu_f(v);
                if (EPI == 3) v += res[(size_t)row * N + col];
                C[(size_t)row * N + col] = v;
            }
        }
    }
}

// ---------------- fused flash attention ----------------
// One wave per (b, head, 16-query tile). qkv: [B*S, 3E]; out: [B*S, E].
__global__ __launch_bounds__(32) void attn_kernel(const float* __restrict__ qkv,
                                                  const int* __restrict__ mask,
                                                  float* __restrict__ out) {
    __shared__ __attribute__((aligned(16))) unsigned short Qs[16 * 72];
    __shared__ __attribute__((aligned(16))) unsigned short Ks[32 * 72];
    __shared__ __attribute__((aligned(16))) unsigned short Vt[64 * 40];
    __shared__ __attribute__((aligned(16))) unsigned short Ps[16 * 40];

    const int qt   = blockIdx.x & 31;          // S/16 = 32 query tiles
    const int bh   = blockIdx.x >> 5;
    const int head = bh % HDIM;
    const int b    = bh / HDIM;
    const int q0   = qt * 16;

    const int lane = threadIdx.x;
    const int half = lane >> 4;
    const int l15  = lane & 15;

    // stage Q tile [16 x 64] -> bf16 LDS
    for (int idx = lane; idx < 16 * 64; idx += 32) {
        int r = idx >> 6, c = idx & 63;
        Qs[r * 72 + c] = f2bf(qkv[(size_t)(b * SDIM + q0 + r) * (3 * EDIM) + head * DH + c]);
    }
    __syncthreads();
    v16bf qa0 = ldfrag(&Qs[l15 * 72 + half * 8],      &Qs[l15 * 72 + 16 + half * 8]);
    v16bf qa1 = ldfrag(&Qs[l15 * 72 + 32 + half * 8], &Qs[l15 * 72 + 48 + half * 8]);

    const v8f vz = {0.f, 0.f, 0.f, 0.f, 0.f, 0.f, 0.f, 0.f};
    v8f o0 = vz, o1 = vz, o2 = vz, o3 = vz;
    float mrow[8], lrow[8];
    #pragma unroll
    for (int r = 0; r < 8; ++r) { mrow[r] = -1e30f; lrow[r] = 0.f; }

    for (int j0 = 0; j0 < q0 + 16; j0 += 32) {     // causal: key tiles up to query tile
        // stage K tile [32 keys x 64]
        for (int idx = lane; idx < 32 * 64; idx += 32) {
            int r = idx >> 6, c = idx & 63;
            Ks[r * 72 + c] = f2bf(qkv[(size_t)(b * SDIM + j0 + r) * (3 * EDIM) + EDIM + head * DH + c]);
        }
        // stage V transposed: Vt[dim][key]
        for (int idx = lane; idx < 64 * 32; idx += 32) {
            int d = idx >> 5, kk = idx & 31;
            Vt[d * 40 + kk] = f2bf(qkv[(size_t)(b * SDIM + j0 + kk) * (3 * EDIM) + 2 * EDIM + head * DH + d]);
        }
        __syncthreads();

        // scores: S[16q, 32k] = Q @ K^T  (two 16-key sub-tiles, K-dim 64 = 2 WMMAs each)
        v8f s0 = vz, s1 = vz;
        {
            v16bf kb00 = ldfrag(&Ks[l15 * 72 + half * 8],        &Ks[l15 * 72 + 16 + half * 8]);
            v16bf kb01 = ldfrag(&Ks[l15 * 72 + 32 + half * 8],   &Ks[l15 * 72 + 48 + half * 8]);
            v16bf kb10 = ldfrag(&Ks[(16 + l15) * 72 + half * 8], &Ks[(16 + l15) * 72 + 16 + half * 8]);
            v16bf kb11 = ldfrag(&Ks[(16 + l15) * 72 + 32 + half * 8], &Ks[(16 + l15) * 72 + 48 + half * 8]);
            s0 = wmma_bf16(qa0, kb00, s0); s0 = wmma_bf16(qa1, kb01, s0);
            s1 = wmma_bf16(qa0, kb10, s1); s1 = wmma_bf16(qa1, kb11, s1);
        }

        const int key0 = j0 + l15, key1 = j0 + 16 + l15;
        const int km0 = mask[b * SDIM + key0];
        const int km1 = mask[b * SDIM + key1];

        #pragma unroll
        for (int r = 0; r < 8; ++r) {
            int qrow = q0 + half * 8 + r;
            float v0 = s0[r] * 0.125f;     // 1/sqrt(64)
            float v1 = s1[r] * 0.125f;
            if (key0 > qrow || km0 == 0) v0 = -1e9f;
            if (key1 > qrow || km1 == 0) v1 = -1e9f;

            // row max across the 16 lanes of this half-group
            float rm = fmaxf(v0, v1);
            for (int x = 1; x < 16; x <<= 1) rm = fmaxf(rm, __shfl_xor(rm, x, 32));
            float mnew = fmaxf(mrow[r], rm);
            float corr = __expf(mrow[r] - mnew);
            float p0 = __expf(v0 - mnew);
            float p1 = __expf(v1 - mnew);
            float rs = p0 + p1;
            for (int x = 1; x < 16; x <<= 1) rs += __shfl_xor(rs, x, 32);
            lrow[r] = lrow[r] * corr + rs;
            mrow[r] = mnew;
            o0[r] *= corr; o1[r] *= corr; o2[r] *= corr; o3[r] *= corr;

            Ps[(half * 8 + r) * 40 + l15]      = f2bf(p0);
            Ps[(half * 8 + r) * 40 + 16 + l15] = f2bf(p1);
        }
        __syncthreads();

        // O += P[16,32] @ V[32,64]
        v16bf pa  = ldfrag(&Ps[l15 * 40 + half * 8], &Ps[l15 * 40 + 16 + half * 8]);
        v16bf vb0 = ldfrag(&Vt[(0  + l15) * 40 + half * 8], &Vt[(0  + l15) * 40 + 16 + half * 8]);
        v16bf vb1 = ldfrag(&Vt[(16 + l15) * 40 + half * 8], &Vt[(16 + l15) * 40 + 16 + half * 8]);
        v16bf vb2 = ldfrag(&Vt[(32 + l15) * 40 + half * 8], &Vt[(32 + l15) * 40 + 16 + half * 8]);
        v16bf vb3 = ldfrag(&Vt[(48 + l15) * 40 + half * 8], &Vt[(48 + l15) * 40 + 16 + half * 8]);
        o0 = wmma_bf16(pa, vb0, o0);
        o1 = wmma_bf16(pa, vb1, o1);
        o2 = wmma_bf16(pa, vb2, o2);
        o3 = wmma_bf16(pa, vb3, o3);
        __syncthreads();
    }

    // normalize and write [B*S, E]
    #pragma unroll
    for (int r = 0; r < 8; ++r) {
        float inv = (lrow[r] > 0.f) ? (1.0f / lrow[r]) : 0.f;
        int row = q0 + half * 8 + r;
        size_t base = (size_t)(b * SDIM + row) * EDIM + head * DH;
        out[base + 0  + l15] = o0[r] * inv;
        out[base + 16 + l15] = o1[r] * inv;
        out[base + 32 + l15] = o2[r] * inv;
        out[base + 48 + l15] = o3[r] * inv;
    }
}

// ---------------- host orchestration ----------------
extern "C" void kernel_launch(void* const* d_in, const int* in_sizes, int n_in,
                              void* d_out, int out_size, void* d_ws, size_t ws_size,
                              hipStream_t stream) {
    (void)in_sizes; (void)n_in; (void)out_size; (void)ws_size;
    const int*   x       = (const int*)  d_in[0];
    const int*   mask    = (const int*)  d_in[1];
    const float* embed_W = (const float*)d_in[2];
    const float* art_W   = (const float*)d_in[3];
    // d_in[4] = sum_W: unused (S=512 <= 514 split)
    const float* attn_w  = (const float*)d_in[5];
    const float* attn_b  = (const float*)d_in[6];
    const float* proj_w  = (const float*)d_in[7];
    const float* proj_b  = (const float*)d_in[8];
    const float* fc_w    = (const float*)d_in[9];
    const float* fc_b    = (const float*)d_in[10];
    const float* mproj_w = (const float*)d_in[11];
    const float* mproj_b = (const float*)d_in[12];
    const float* ln1_g   = (const float*)d_in[13];
    const float* ln1_b   = (const float*)d_in[14];
    const float* ln2_g   = (const float*)d_in[15];
    const float* ln2_b   = (const float*)d_in[16];
    float* logits = (float*)d_out;

    // scratch layout (floats)
    float* h    = (float*)d_ws;                         // [M, E]
    float* qkv  = h    + (size_t)MROWS * EDIM;          // [M, 3E]
    float* attn = qkv  + (size_t)MROWS * 3 * EDIM;      // [M, E]
    float* rb   = attn + (size_t)MROWS * EDIM;          // [M, E]
    float* n1   = rb   + (size_t)MROWS * EDIM;          // [M, E]
    float* ff   = n1   + (size_t)MROWS * EDIM;          // [M, FF]

    embed_kernel<<<MROWS, 192, 0, stream>>>(x, mask, embed_W, art_W, h);

    const dim3 blk(256);
    for (int l = 0; l < LDIM; ++l) {
        const float* aw = attn_w  + (size_t)l * EDIM * 3 * EDIM;
        const float* ab = attn_b  + (size_t)l * 3 * EDIM;
        const float* pw = proj_w  + (size_t)l * EDIM * EDIM;
        const float* pb = proj_b  + (size_t)l * EDIM;
        const float* fw = fc_w    + (size_t)l * EDIM * FFDIM;
        const float* fb = fc_b    + (size_t)l * FFDIM;
        const float* mw = mproj_w + (size_t)l * FFDIM * EDIM;
        const float* mb = mproj_b + (size_t)l * EDIM;
        const float* g1 = ln1_g + (size_t)l * EDIM;
        const float* b1 = ln1_b + (size_t)l * EDIM;
        const float* g2 = ln2_g + (size_t)l * EDIM;
        const float* b2 = ln2_b + (size_t)l * EDIM;

        // qkv = h @ aw + ab
        gemm_kernel<1, false><<<dim3(3 * EDIM / 128, MROWS / 128), blk, 0, stream>>>(
            h, aw, ab, nullptr, qkv, MROWS, 3 * EDIM, EDIM);
        // fused attention
        attn_kernel<<<BDIM * HDIM * (SDIM / 16), 32, 0, stream>>>(qkv, mask, attn);
        // rb = attn @ pw + pb + h
        gemm_kernel<3, false><<<dim3(EDIM / 128, MROWS / 128), blk, 0, stream>>>(
            attn, pw, pb, h, rb, MROWS, EDIM, EDIM);
        // n1 = LN1(rb)
        ln_kernel<<<MROWS, 256, 0, stream>>>(rb, g1, b1, n1);
        // ff = gelu(n1 @ fw + fb)
        gemm_kernel<2, false><<<dim3(FFDIM / 128, MROWS / 128), blk, 0, stream>>>(
            n1, fw, fb, nullptr, ff, MROWS, FFDIM, EDIM);
        // rb = ff @ mw + mb + n1
        gemm_kernel<3, false><<<dim3(EDIM / 128, MROWS / 128), blk, 0, stream>>>(
            ff, mw, mb, n1, rb, MROWS, EDIM, FFDIM);
        // h = LN2(rb)
        ln_kernel<<<MROWS, 256, 0, stream>>>(rb, g2, b2, h);
    }

    // logits = h @ embed_W^T   (B stored [V, E] row-major -> BT mode)
    gemm_kernel<0, true><<<dim3((VDIM + 127) / 128, MROWS / 128), blk, 0, stream>>>(
        h, embed_W, nullptr, nullptr, logits, MROWS, VDIM, EDIM);
}
